// CausalSelfAttention_4681514353257
// MI455X (gfx1250) — matrix operand order repaired and checked by reading, not compiled, not run on hardware
//
#include <hip/hip_runtime.h>
#include <hip/hip_bf16.h>
#include <stdint.h>

// Problem constants (match reference)
#define BATCH  4
#define SEQ    2048
#define DMODEL 1024
#define NHEADS 16
#define DHEAD  64
#define MROWS  (BATCH * SEQ)   // 8192

typedef __bf16 bf16_t;
typedef __bf16 v16bf __attribute__((ext_vector_type(16)));
typedef __bf16 v8bf  __attribute__((ext_vector_type(8)));
typedef float  v8f   __attribute__((ext_vector_type(8)));

static __device__ __forceinline__ bf16_t f2bf(float f) {
  uint32_t u = __float_as_uint(f);
  u += 0x7FFFu + ((u >> 16) & 1u);   // round-to-nearest-even
  uint16_t h = (uint16_t)(u >> 16);
  bf16_t r;
  __builtin_memcpy(&r, &h, 2);
  return r;
}

// ---- WMMA fragment loaders (per CDNA5 ISA 7.12.2 layouts) ----
// 16-bit A matrix 16x32: lane L holds row m=L&15; K-pairs form two contiguous
// 16B runs: K = half*8..half*8+7 and 16+half*8..16+half*8+7  (half = L>>4).
static __device__ __forceinline__ v16bf load_a_frag(const bf16_t* base, int ld) {
  int lane = threadIdx.x & 31;
  int m = lane & 15;
  int half = lane >> 4;
  const bf16_t* row = base + m * ld + half * 8;
  v8bf lo = *(const v8bf*)(row);
  v8bf hi = *(const v8bf*)(row + 16);
  v16bf out;
#pragma unroll
  for (int i = 0; i < 8; ++i) { out[i] = lo[i]; out[i + 8] = hi[i]; }
  return out;
}

// 16-bit B matrix 32x16: lane L holds column n=L&15; K = half*8..half*8+7 and
// 16+half*8..16+half*8+7 (half = L>>4), matching the A fragment order.
static __device__ __forceinline__ v16bf load_b_frag(const bf16_t* base, int ld) {
  int lane = threadIdx.x & 31;
  int n = lane & 15;
  int half = lane >> 4;
  v16bf out;
#pragma unroll
  for (int i = 0; i < 8; ++i) {
    out[i] = base[(half * 8 + i) * ld + n];
    out[i + 8] = base[(16 + half * 8 + i) * ld + n];
  }
  return out;
}

static __device__ __forceinline__ v8f wmma_bf16(v16bf a, v16bf b, v8f c) {
  return __builtin_amdgcn_wmma_f32_16x16x32_bf16(false, a, false, b, (short)0, c,
                                                 false, false);
}

// ---- conversion kernels ----
// Vectorized f32 -> bf16 (4 elements/thread: b128 read, b64 write)
__global__ void f32_to_bf16_kernel(const float4* __restrict__ in,
                                   bf16_t* __restrict__ out, int n4) {
  int i = blockIdx.x * blockDim.x + threadIdx.x;
  int stride = gridDim.x * blockDim.x;
  for (; i < n4; i += stride) {
    float4 v = in[i];
    bf16_t* o = out + (size_t)i * 4;
    o[0] = f2bf(v.x); o[1] = f2bf(v.y); o[2] = f2bf(v.z); o[3] = f2bf(v.w);
  }
}

// LDS-tiled transpose: Wt[k*D + n] = W[n*D + k], coalesced both sides.
// Grid: (D/32, D/32), 256 threads (32x8).
__global__ __launch_bounds__(256) void transpose_w_kernel(
    const float* __restrict__ W, bf16_t* __restrict__ Wt) {
  __shared__ bf16_t tile[32][33];  // pad to dodge bank conflicts
  int tx = threadIdx.x & 31, ty = threadIdx.x >> 5;   // 32 x 8
  int kb = blockIdx.x * 32;  // k-range of this tile
  int nb = blockIdx.y * 32;  // n-range of this tile
#pragma unroll
  for (int j = 0; j < 4; ++j) {
    int n = nb + ty + j * 8;
    tile[ty + j * 8][tx] = f2bf(W[(size_t)n * DMODEL + kb + tx]);  // coalesced read
  }
  __syncthreads();
#pragma unroll
  for (int j = 0; j < 4; ++j) {
    int k = kb + ty + j * 8;
    Wt[(size_t)k * DMODEL + nb + tx] = tile[tx][ty + j * 8];       // coalesced write
  }
}

// ---- bf16 GEMM: C[M=8192][N] = A[M][K] * B[K][N] ----
// 256 threads = 8 waves; block tile 128x128, k-step 32, double-buffered LDS.
// Waves: 4(m) x 2(n); each wave owns 32x64 = 2x4 WMMA tiles.
// out_mode 0: fp32 plain row-major.  out_mode 1: bf16 in [B,H,T,dh] layout.
__global__ __launch_bounds__(256) void gemm_bf16_kernel(
    const bf16_t* __restrict__ A, const bf16_t* __restrict__ Bm,
    float* __restrict__ Cf, bf16_t* __restrict__ Cb,
    int Kdim, int Ndim, int out_mode, float scale) {
  __shared__ bf16_t As[2][128 * 32];
  __shared__ bf16_t Bs[2][32 * 128];

  int tid = threadIdx.x;
  int lane = tid & 31;
  int w = tid >> 5;
  int wm = w & 3, wn = w >> 2;
  int m0 = blockIdx.y * 128, n0 = blockIdx.x * 128;

  v8f zero = {0.f, 0.f, 0.f, 0.f, 0.f, 0.f, 0.f, 0.f};
  v8f acc[2][4];
#pragma unroll
  for (int i = 0; i < 2; ++i)
#pragma unroll
    for (int n = 0; n < 4; ++n) acc[i][n] = zero;

  int arow = tid >> 1, acol = (tid & 1) * 16;   // 128 rows x 32 cols
  int brow = tid >> 3, bcol = (tid & 7) * 16;   // 32 rows x 128 cols
  const bf16_t* aptr = A + (size_t)(m0 + arow) * Kdim + acol;
  const bf16_t* bptr = Bm + (size_t)brow * Ndim + n0 + bcol;

  // prologue: stage tile 0
  {
    v8bf a0 = *(const v8bf*)(aptr);
    v8bf a1 = *(const v8bf*)(aptr + 8);
    v8bf b0 = *(const v8bf*)(bptr);
    v8bf b1 = *(const v8bf*)(bptr + 8);
    *(v8bf*)&As[0][arow * 32 + acol]      = a0;
    *(v8bf*)&As[0][arow * 32 + acol + 8]  = a1;
    *(v8bf*)&Bs[0][brow * 128 + bcol]     = b0;
    *(v8bf*)&Bs[0][brow * 128 + bcol + 8] = b1;
  }
  __syncthreads();

  int buf = 0;
  for (int k0 = 0; k0 < Kdim; k0 += 32) {
    bool more = (k0 + 32) < Kdim;
    // issue next tile's global loads early (overlap with WMMA compute)
    v8bf na0, na1, nb0, nb1;
    if (more) {
      const bf16_t* an = aptr + (k0 + 32);
      const bf16_t* bn = bptr + (size_t)(k0 + 32) * Ndim;
      na0 = *(const v8bf*)(an);
      na1 = *(const v8bf*)(an + 8);
      nb0 = *(const v8bf*)(bn);
      nb1 = *(const v8bf*)(bn + 8);
    }

    const bf16_t* Ab = As[buf];
    const bf16_t* Bb = Bs[buf];
    v16bf fa0 = load_a_frag(&Ab[(wm * 32 + 0) * 32], 32);
    v16bf fa1 = load_a_frag(&Ab[(wm * 32 + 16) * 32], 32);
    v16bf fb[4];
#pragma unroll
    for (int n = 0; n < 4; ++n) fb[n] = load_b_frag(&Bb[wn * 64 + n * 16], 128);
#pragma unroll
    for (int n = 0; n < 4; ++n) {
      acc[0][n] = wmma_bf16(fa0, fb[n], acc[0][n]);
      acc[1][n] = wmma_bf16(fa1, fb[n], acc[1][n]);
    }

    if (more) {
      int nb2 = buf ^ 1;
      *(v8bf*)&As[nb2][arow * 32 + acol]      = na0;
      *(v8bf*)&As[nb2][arow * 32 + acol + 8]  = na1;
      *(v8bf*)&Bs[nb2][brow * 128 + bcol]     = nb0;
      *(v8bf*)&Bs[nb2][brow * 128 + bcol + 8] = nb1;
    }
    __syncthreads();
    buf ^= 1;
  }

  int half = lane >> 4, l15 = lane & 15;
#pragma unroll
  for (int i = 0; i < 2; ++i)
#pragma unroll
    for (int n = 0; n < 4; ++n)
#pragma unroll
      for (int r = 0; r < 8; ++r) {
        int m = m0 + wm * 32 + i * 16 + half * 8 + r;
        int nc = n0 + wn * 64 + n * 16 + l15;
        float v = acc[i][n][r] * scale;
        if (out_mode == 0) {
          Cf[(size_t)m * Ndim + nc] = v;
        } else {
          int b = m >> 11, t = m & (SEQ - 1);
          int h = nc >> 6, d = nc & (DHEAD - 1);
          Cb[(((size_t)(b * NHEADS + h)) * SEQ + t) * DHEAD + d] = f2bf(v);
        }
      }
}

// ---- flash attention (causal), Q pre-scaled by 1/sqrt(dh) ----
// Grid: (SEQ/128, BATCH*NHEADS). 256 threads = 8 waves, each wave: 16 q-rows.
// Key blocks of 64 processed cooperatively (Kt transposed + Vs in LDS).
__global__ __launch_bounds__(256) void flash_attn_kernel(
    const bf16_t* __restrict__ Qg, const bf16_t* __restrict__ Kg,
    const bf16_t* __restrict__ Vg, bf16_t* __restrict__ Yb) {
  __shared__ bf16_t Kt[64 * 64];       // [d][key]
  __shared__ bf16_t Vs[64 * 64];       // [key][d]
  __shared__ bf16_t Ps[8][16 * 64];    // per-wave P staging, [q][key]

  int tid = threadIdx.x;
  int lane = tid & 31;
  int w = tid >> 5;
  int half = lane >> 4, l15 = lane & 15;

  int bh = blockIdx.y;
  int b = bh >> 4, h = bh & (NHEADS - 1);
  int q0 = blockIdx.x * 128 + w * 16;  // this wave's first query row (t index)

  // Q fragments stay in registers for the whole kernel (16x64 = 2 A-frags)
  const bf16_t* qbase = Qg + ((size_t)bh * SEQ + q0) * DHEAD;
  v16bf aq0 = load_a_frag(qbase, DHEAD);        // d = 0..31
  v16bf aq1 = load_a_frag(qbase + 32, DHEAD);   // d = 32..63

  v8f zero = {0.f, 0.f, 0.f, 0.f, 0.f, 0.f, 0.f, 0.f};
  v8f o[4] = {zero, zero, zero, zero};
  float mrow[8], lrow[8];
#pragma unroll
  for (int r = 0; r < 8; ++r) { mrow[r] = -1e30f; lrow[r] = 0.f; }

  int nkb = 2 * (blockIdx.x + 1);  // key blocks of 64, up to causal frontier
  for (int kb = 0; kb < nkb; ++kb) {
    int k0 = kb * 64;
    __syncthreads();  // prior iteration done reading Kt/Vs
    {
      int key = tid >> 2;            // 0..63
      int dbase = (tid & 3) * 16;    // 0,16,32,48
      const bf16_t* kg = Kg + ((size_t)bh * SEQ + k0 + key) * DHEAD + dbase;
      v8bf ka = *(const v8bf*)kg;
      v8bf kbv = *(const v8bf*)(kg + 8);
#pragma unroll
      for (int j = 0; j < 8; ++j) {
        Kt[(dbase + j) * 64 + key]     = ka[j];
        Kt[(dbase + 8 + j) * 64 + key] = kbv[j];
      }
      const bf16_t* vg = Vg + ((size_t)bh * SEQ + k0 + key) * DHEAD + dbase;
      *(v8bf*)&Vs[key * 64 + dbase]     = *(const v8bf*)vg;
      *(v8bf*)&Vs[key * 64 + dbase + 8] = *(const v8bf*)(vg + 8);
    }
    __syncthreads();

    if (k0 <= q0 + 15) {  // wave has at least one unmasked element
      // S = Q * K^T  (fp32 accum), 4 n-tiles of 16 keys
      v8f s[4];
#pragma unroll
      for (int n = 0; n < 4; ++n) {
        s[n] = zero;
        s[n] = wmma_bf16(aq0, load_b_frag(&Kt[0 * 64 + n * 16], 64), s[n]);
        s[n] = wmma_bf16(aq1, load_b_frag(&Kt[32 * 64 + n * 16], 64), s[n]);
      }
      // causal mask
#pragma unroll
      for (int n = 0; n < 4; ++n)
#pragma unroll
        for (int r = 0; r < 8; ++r) {
          int qrow = q0 + half * 8 + r;
          int key = k0 + n * 16 + l15;
          if (key > qrow) s[n][r] = -1e30f;
        }
      // online softmax: row max (across 4 tiles + 16 lanes of this half)
      float mnew[8], alpha[8];
#pragma unroll
      for (int r = 0; r < 8; ++r) {
        float v = fmaxf(fmaxf(s[0][r], s[1][r]), fmaxf(s[2][r], s[3][r]));
        v = fmaxf(v, __shfl_xor(v, 1));
        v = fmaxf(v, __shfl_xor(v, 2));
        v = fmaxf(v, __shfl_xor(v, 4));
        v = fmaxf(v, __shfl_xor(v, 8));
        mnew[r] = fmaxf(mrow[r], v);
        alpha[r] = __expf(mrow[r] - mnew[r]);
        mrow[r] = mnew[r];
      }
      // P = exp(S - m), stage to LDS (C-layout -> row-major [16][64])
      float rs[8];
#pragma unroll
      for (int r = 0; r < 8; ++r) rs[r] = 0.f;
      bf16_t* myP = &Ps[w][0];
#pragma unroll
      for (int n = 0; n < 4; ++n)
#pragma unroll
        for (int r = 0; r < 8; ++r) {
          float sv = s[n][r];
          float p = (sv > -1e29f) ? __expf(sv - mnew[r]) : 0.f;
          rs[r] += p;
          myP[(half * 8 + r) * 64 + n * 16 + l15] = f2bf(p);
        }
#pragma unroll
      for (int r = 0; r < 8; ++r) {
        float t = rs[r];
        t += __shfl_xor(t, 1);
        t += __shfl_xor(t, 2);
        t += __shfl_xor(t, 4);
        t += __shfl_xor(t, 8);
        lrow[r] = lrow[r] * alpha[r] + t;
      }
      // rescale O, then O += P * V
#pragma unroll
      for (int nd = 0; nd < 4; ++nd)
#pragma unroll
        for (int r = 0; r < 8; ++r) o[nd][r] *= alpha[r];

      asm volatile("s_wait_dscnt 0" ::: "memory");
      v16bf ap0 = load_a_frag(myP, 64);        // keys 0..31
      v16bf ap1 = load_a_frag(myP + 32, 64);   // keys 32..63
#pragma unroll
      for (int nd = 0; nd < 4; ++nd) {
        o[nd] = wmma_bf16(ap0, load_b_frag(&Vs[0 * 64 + nd * 16], 64), o[nd]);
        o[nd] = wmma_bf16(ap1, load_b_frag(&Vs[32 * 64 + nd * 16], 64), o[nd]);
      }
    }
  }

  // normalize and write back in [B,T,D] bf16 layout for the Wo GEMM
  float invl[8];
#pragma unroll
  for (int r = 0; r < 8; ++r) invl[r] = (lrow[r] > 0.f) ? 1.f / lrow[r] : 0.f;
#pragma unroll
  for (int nd = 0; nd < 4; ++nd)
#pragma unroll
    for (int r = 0; r < 8; ++r) {
      int t = q0 + half * 8 + r;
      int col = h * DHEAD + nd * 16 + l15;
      Yb[((size_t)b * SEQ + t) * DMODEL + col] = f2bf(o[nd][r] * invl[r]);
    }
}

extern "C" void kernel_launch(void* const* d_in, const int* in_sizes, int n_in,
                              void* d_out, int out_size, void* d_ws, size_t ws_size,
                              hipStream_t stream) {
  (void)in_sizes; (void)n_in; (void)out_size; (void)ws_size;
  const float* x  = (const float*)d_in[0];
  const float* Wq = (const float*)d_in[1];
  const float* Wk = (const float*)d_in[2];
  const float* Wv = (const float*)d_in[3];
  const float* Wo = (const float*)d_in[4];
  float* out = (float*)d_out;

  char* ws = (char*)d_ws;
  size_t off = 0;
  auto alloc = [&](size_t bytes) -> void* {
    void* p = ws + off;
    off = (off + bytes + 255) & ~(size_t)255;
    return p;
  };
  bf16_t* xb  = (bf16_t*)alloc((size_t)MROWS * DMODEL * 2);   // 16 MB
  bf16_t* WqT = (bf16_t*)alloc((size_t)DMODEL * DMODEL * 2);  // 2 MB
  bf16_t* WkT = (bf16_t*)alloc((size_t)DMODEL * DMODEL * 2);
  bf16_t* WvT = (bf16_t*)alloc((size_t)DMODEL * DMODEL * 2);
  bf16_t* WoT = (bf16_t*)alloc((size_t)DMODEL * DMODEL * 2);
  bf16_t* Qh  = (bf16_t*)alloc((size_t)MROWS * DMODEL * 2);   // [B,H,T,dh]
  bf16_t* Kh  = (bf16_t*)alloc((size_t)MROWS * DMODEL * 2);
  bf16_t* Vh  = (bf16_t*)alloc((size_t)MROWS * DMODEL * 2);
  bf16_t* Yb  = (bf16_t*)alloc((size_t)MROWS * DMODEL * 2);   // attention out

  // 1) convert inputs (vectorized) + tiled weight transposes
  f32_to_bf16_kernel<<<2048, 256, 0, stream>>>((const float4*)x, xb,
                                               MROWS * DMODEL / 4);
  dim3 tgrid(DMODEL / 32, DMODEL / 32);
  transpose_w_kernel<<<tgrid, 256, 0, stream>>>(Wq, WqT);
  transpose_w_kernel<<<tgrid, 256, 0, stream>>>(Wk, WkT);
  transpose_w_kernel<<<tgrid, 256, 0, stream>>>(Wv, WvT);
  transpose_w_kernel<<<tgrid, 256, 0, stream>>>(Wo, WoT);

  // 2) Q/K/V projections -> [B,H,T,dh] bf16 (Q pre-scaled by 1/sqrt(dh))
  dim3 ggrid(DMODEL / 128, MROWS / 128);
  gemm_bf16_kernel<<<ggrid, 256, 0, stream>>>(xb, WqT, nullptr, Qh,
                                              DMODEL, DMODEL, 1, 0.125f);
  gemm_bf16_kernel<<<ggrid, 256, 0, stream>>>(xb, WkT, nullptr, Kh,
                                              DMODEL, DMODEL, 1, 1.0f);
  gemm_bf16_kernel<<<ggrid, 256, 0, stream>>>(xb, WvT, nullptr, Vh,
                                              DMODEL, DMODEL, 1, 1.0f);

  // 3) causal flash attention
  dim3 agrid(SEQ / 128, BATCH * NHEADS);
  flash_attn_kernel<<<agrid, 256, 0, stream>>>(Qh, Kh, Vh, Yb);

  // 4) output projection -> fp32 d_out
  gemm_bf16_kernel<<<ggrid, 256, 0, stream>>>(Yb, WoT, out, nullptr,
                                              DMODEL, DMODEL, 0, 1.0f);
}
